// Expert_61942018343437
// MI455X (gfx1250) — compile-verified, matrix operand order checked by reading, not run on hardware
//
#include <hip/hip_runtime.h>

typedef unsigned short ushort_t;
typedef __attribute__((ext_vector_type(16))) __bf16 bf16x16;
typedef __attribute__((ext_vector_type(8)))  float   f32x8;
typedef int v4i_vs __attribute__((vector_size(16)));
typedef __attribute__((ext_vector_type(4))) unsigned int u32x4;
typedef __attribute__((ext_vector_type(8))) int i32x8;
typedef __attribute__((ext_vector_type(4))) int i32x4;

#if defined(__has_builtin)
#if __has_builtin(__builtin_amdgcn_tensor_load_to_lds) && \
    __has_builtin(__builtin_amdgcn_s_wait_tensorcnt)
#define USE_TDM 1
#endif
#if __has_builtin(__builtin_amdgcn_global_load_async_to_lds_b128) && \
    __has_builtin(__builtin_amdgcn_s_wait_asynccnt)
#define USE_ASYNC_LDS 1
#endif
#endif

constexpr int E = 8, T = 2048, H = 4096, F = 2048;

constexpr int BM  = 128;            // rows (tokens) per block tile
constexpr int BK  = 32;             // K per LDS tile (== WMMA K)
constexpr int LDA = BK * 2 + 16;    // 80 B padded row stride (conflict-free)
constexpr int LDB = BK * 2 + 16;    // 80 B padded col stride
constexpr int BCOLS  = 128;         // LDS B columns per tile
constexpr int ABYTES = BM * LDA;    // 10240
constexpr int BBYTES = BCOLS * LDB; // 10240
constexpr int BUF    = ABYTES + BBYTES; // 20480 per buffer; x2 = 40 KB LDS

struct Chunk32 { uint4 a, b; };

__device__ __forceinline__ bf16x16 ldsFrag(const unsigned char* base) {
  Chunk32 c;
  c.a = *(const uint4*)(base);
  c.b = *(const uint4*)(base + 32);
  return __builtin_bit_cast(bf16x16, c);
}

__device__ __forceinline__ ushort_t f2bf(float f) {
  unsigned u = __builtin_bit_cast(unsigned, f);
  u += 0x7FFFu + ((u >> 16) & 1u);          // round-to-nearest-even
  return (ushort_t)(u >> 16);
}

// Branchless GELU(tanh): 0.5*x*(1+tanh(u)) == x * sigmoid(2u).
__device__ __forceinline__ float gelu_tanh(float x) {
  float u  = 0.7978845608028654f * (x + 0.044715f * x * x * x);
  float ev = __expf(-2.0f * u);
  return x * __builtin_amdgcn_rcpf(1.0f + ev);
}

#if defined(USE_ASYNC_LDS)
// Direct global -> LDS 16B async copy (ASYNCcnt-tracked, no VGPR round trip).
__device__ __forceinline__ void asyncCopy16(const void* g, void* lds) {
  __builtin_amdgcn_global_load_async_to_lds_b128(
      (__attribute__((address_space(1))) v4i_vs*)(uintptr_t)g,
      (__attribute__((address_space(3))) v4i_vs*)(unsigned)(uintptr_t)lds, 0, 0);
}
#endif

#if defined(USE_TDM)
// TDM: one 2-D tile DMA, global(row stride ldK elems) -> LDS with 16B row pad
// (pad_interval=16 DW per row of 64 B, pad_amount=4 DW) => LDS stride == LDA.
__device__ __forceinline__ void tdmLoadTile(const void* gtile, void* ldsDst,
                                            unsigned ldK, unsigned nRows) {
  unsigned long long ga = (unsigned long long)(uintptr_t)gtile;
  unsigned lds = (unsigned)(uintptr_t)ldsDst;
  u32x4 g0;
  g0[0] = 1u;                               // count=1 valid descriptor
  g0[1] = lds;                              // lds_addr
  g0[2] = (unsigned)ga;                     // global_addr[31:0]
  g0[3] = (unsigned)((ga >> 32) & 0x1FFFFFFu) | (2u << 30);  // addr[56:32]|type=2
  i32x8 g1;
  g1[0] = (int)((1u << 16) |                // data_size = 2 bytes
                (1u << 20) |                // pad_enable
                (3u << 22) |                // pad_interval: 16 DWORDs (64 B row)
                (3u << 25));                // pad_amount: 4 DWORDs (16 B)
  g1[1] = (int)((ldK & 0xFFFFu) << 16);                         // tensor_dim0 lo
  g1[2] = (int)(((ldK >> 16) & 0xFFFFu) | ((nRows & 0xFFFFu) << 16)); // dim0 hi|dim1 lo
  g1[3] = (int)(((nRows >> 16) & 0xFFFFu) | ((unsigned)BK << 16));    // dim1 hi|tile_dim0
  g1[4] = (int)BM;                          // tile_dim1 = 128 rows, tile_dim2 = 0
  g1[5] = (int)ldK;                         // tensor_dim0_stride (elements)
  g1[6] = 0;
  g1[7] = 0;
  i32x4 z4 = {0, 0, 0, 0};
  i32x8 z8 = {0, 0, 0, 0, 0, 0, 0, 0};
  __builtin_amdgcn_tensor_load_to_lds(g0, g1, z4, z4, z8, 0);
}
#endif

// ---------------------------------------------------------------------------
// Kernel 1: per expert, C[T, 2F] = X[T,H] @ W1[H,2F]; emit g = gelu(a)*b
// ---------------------------------------------------------------------------
__global__ __launch_bounds__(256) void moe_fc1_glu(const ushort_t* __restrict__ X,
                                                   const ushort_t* __restrict__ W1,
                                                   ushort_t* __restrict__ G) {
  __shared__ alignas(16) unsigned char smem[2 * BUF];

  const int e  = blockIdx.z;
  const int m0 = blockIdx.x * BM;
  const int n0 = blockIdx.y * 64;

  const ushort_t* Xe  = X  + (size_t)e * T * H;
  const ushort_t* W1e = W1 + (size_t)e * H * (2 * F);
  ushort_t*       Ge  = G  + (size_t)e * T * F;

  const int tid  = threadIdx.x;
  const int lane = tid & 31;
  const int wave = tid >> 5;
  const int wm   = wave & 3;
  const int wn   = wave >> 2;
  const int hs   = lane >> 4;
  const int l15  = lane & 15;

  const int ar0 = tid >> 2;
  const int aq0 = tid & 3;
  const int ar1 = ar0 + 64;
  const int kp  = tid >> 4;
  const int cg  = tid & 15;
  const int c0  = cg * 8;
  const size_t gc0 = (cg < 8) ? (size_t)(n0 + c0)
                              : (size_t)(F + n0 + (c0 - 64));

  f32x8 accA[2][2] = {};
  f32x8 accB[2][2] = {};

  uint4 rb0, rb1;
#if !defined(USE_TDM) && !defined(USE_ASYNC_LDS)
  uint4 ra0v, ra1v;
#endif
  auto fetchB = [&](int kt) {
    const ushort_t* bk = W1e + (size_t)(kt * BK + kp * 2) * (2 * F) + gc0;
    rb0 = *(const uint4*)(bk);
    rb1 = *(const uint4*)(bk + 2 * F);
  };
  auto loadA = [&](int kt, int p) {
    const ushort_t* ak = Xe + (size_t)m0 * H + (size_t)kt * BK;
    unsigned char* bufA = smem + p * BUF;
#if defined(USE_TDM)
    if (wave == 0) tdmLoadTile(ak, bufA, H, T);
#elif defined(USE_ASYNC_LDS)
    asyncCopy16(ak + (size_t)ar0 * H + aq0 * 8, bufA + ar0 * LDA + aq0 * 16);
    asyncCopy16(ak + (size_t)ar1 * H + aq0 * 8, bufA + ar1 * LDA + aq0 * 16);
#else
    (void)p;
    ra0v = *(const uint4*)(ak + (size_t)ar0 * H + aq0 * 8);
    ra1v = *(const uint4*)(ak + (size_t)ar1 * H + aq0 * 8);
#endif
  };
#if !defined(USE_TDM) && !defined(USE_ASYNC_LDS)
  auto stageA = [&](int p) {
    unsigned char* bufA = smem + p * BUF;
    *(uint4*)(bufA + ar0 * LDA + aq0 * 16) = ra0v;
    *(uint4*)(bufA + ar1 * LDA + aq0 * 16) = ra1v;
  };
#endif
  auto waitA = [&]() {
#if defined(USE_TDM)
    if (wave == 0) __builtin_amdgcn_s_wait_tensorcnt(0);
#elif defined(USE_ASYNC_LDS)
    __builtin_amdgcn_s_wait_asynccnt(0);
#endif
  };
  auto prefetch = [&](int kt) {
    const ushort_t* ak = Xe + (size_t)m0 * H + (size_t)kt * BK;
    __builtin_prefetch(ak + (size_t)ar0 * H + aq0 * 8, 0, 1);
    __builtin_prefetch(ak + (size_t)ar1 * H + aq0 * 8, 0, 1);
    const ushort_t* bk = W1e + (size_t)(kt * BK + kp * 2) * (2 * F) + gc0;
    __builtin_prefetch(bk, 0, 1);
    __builtin_prefetch(bk + 2 * F, 0, 1);
  };
  auto stageB = [&](int p) {
    unsigned char* bufB = smem + p * BUF + ABYTES;
    const unsigned* r0 = (const unsigned*)&rb0;
    const unsigned* r1 = (const unsigned*)&rb1;
#pragma unroll
    for (int w = 0; w < 4; ++w) {   // one v_perm_b32 per packed (k,k+1) dword
      unsigned lo = __builtin_amdgcn_perm(r1[w], r0[w], 0x05040100u);
      unsigned hi = __builtin_amdgcn_perm(r1[w], r0[w], 0x07060302u);
      *(unsigned*)(bufB + (c0 + 2 * w)     * LDB + kp * 4) = lo;
      *(unsigned*)(bufB + (c0 + 2 * w + 1) * LDB + kp * 4) = hi;
    }
  };

  constexpr int KT = H / BK;
  fetchB(0);
  loadA(0, 0);
#if !defined(USE_TDM) && !defined(USE_ASYNC_LDS)
  stageA(0);
#endif
  stageB(0);
  for (int kt = 0; kt < KT; ++kt) {
    const int p = kt & 1;
    const bool hasNext = (kt + 1) < KT;
    if (hasNext) fetchB(kt + 1);
#if !defined(USE_TDM) && !defined(USE_ASYNC_LDS)
    if (hasNext) loadA(kt + 1, p ^ 1);
#endif
    if (kt + 2 < KT) prefetch(kt + 2);
    waitA();                               // my tile-p DMA/async stores complete
    __syncthreads();                       // publish tile p; readers of p^1 done
    if (hasNext) {
      stageB(p ^ 1);
#if defined(USE_TDM) || defined(USE_ASYNC_LDS)
      loadA(kt + 1, p ^ 1);                // DMA into p^1 (readers are past barrier)
#else
      stageA(p ^ 1);
#endif
    }

    const unsigned char* bufA = smem + p * BUF;
    const unsigned char* bufB = bufA + ABYTES;

    bf16x16 af[2], bfA[2], bfB[2];
#pragma unroll
    for (int mi = 0; mi < 2; ++mi) {
      int row = wm * 32 + mi * 16 + l15;
      af[mi] = ldsFrag(bufA + row * LDA + hs * 16);
    }
#pragma unroll
    for (int ni = 0; ni < 2; ++ni) {
      int colA = wn * 32 + ni * 16 + l15;
      bfA[ni] = ldsFrag(bufB + colA * LDB + hs * 16);
      bfB[ni] = ldsFrag(bufB + (64 + colA) * LDB + hs * 16);
    }
#pragma unroll
    for (int mi = 0; mi < 2; ++mi)
#pragma unroll
      for (int ni = 0; ni < 2; ++ni) {
        accA[mi][ni] = __builtin_amdgcn_wmma_f32_16x16x32_bf16(
            false, af[mi], false, bfA[ni], (short)0, accA[mi][ni], false, false);
        accB[mi][ni] = __builtin_amdgcn_wmma_f32_16x16x32_bf16(
            false, af[mi], false, bfB[ni], (short)0, accB[mi][ni], false, false);
      }
  }

#pragma unroll
  for (int mi = 0; mi < 2; ++mi)
#pragma unroll
    for (int ni = 0; ni < 2; ++ni) {
      int colg = n0 + wn * 32 + ni * 16 + l15;
#pragma unroll
      for (int r = 0; r < 8; ++r) {
        int row = m0 + wm * 32 + mi * 16 + r + 8 * hs;
        float g = gelu_tanh(accA[mi][ni][r]) * accB[mi][ni][r];
        Ge[(size_t)row * F + colg] = f2bf(g);
      }
    }
}

// ---------------------------------------------------------------------------
// Kernel 2: per expert, Y[T,H] = G[T,F] @ W2[F,H].
// ---------------------------------------------------------------------------
__global__ __launch_bounds__(256) void moe_fc2(const ushort_t* __restrict__ Gin,
                                               const ushort_t* __restrict__ W2,
                                               ushort_t* __restrict__ Y) {
  __shared__ alignas(16) unsigned char smem[2 * BUF];

  const int e  = blockIdx.z;
  const int m0 = blockIdx.x * BM;
  const int n0 = blockIdx.y * 128;

  const ushort_t* Ge  = Gin + (size_t)e * T * F;
  const ushort_t* W2e = W2  + (size_t)e * F * H;
  ushort_t*       Ye  = Y   + (size_t)e * T * H;

  const int tid  = threadIdx.x;
  const int lane = tid & 31;
  const int wave = tid >> 5;
  const int wm   = wave & 3;
  const int wn   = wave >> 2;
  const int hs   = lane >> 4;
  const int l15  = lane & 15;

  const int ar0 = tid >> 2;
  const int aq0 = tid & 3;
  const int ar1 = ar0 + 64;
  const int kp  = tid >> 4;
  const int cg  = tid & 15;
  const int c0  = cg * 8;
  const size_t gc0 = (size_t)(n0 + c0);

  f32x8 acc[2][4] = {};

  uint4 rb0, rb1;
#if !defined(USE_TDM) && !defined(USE_ASYNC_LDS)
  uint4 ra0v, ra1v;
#endif
  auto fetchB = [&](int kt) {
    const ushort_t* bk = W2e + (size_t)(kt * BK + kp * 2) * H + gc0;
    rb0 = *(const uint4*)(bk);
    rb1 = *(const uint4*)(bk + H);
  };
  auto loadA = [&](int kt, int p) {
    const ushort_t* ak = Ge + (size_t)m0 * F + (size_t)kt * BK;
    unsigned char* bufA = smem + p * BUF;
#if defined(USE_TDM)
    if (wave == 0) tdmLoadTile(ak, bufA, F, T);
#elif defined(USE_ASYNC_LDS)
    asyncCopy16(ak + (size_t)ar0 * F + aq0 * 8, bufA + ar0 * LDA + aq0 * 16);
    asyncCopy16(ak + (size_t)ar1 * F + aq0 * 8, bufA + ar1 * LDA + aq0 * 16);
#else
    (void)p;
    ra0v = *(const uint4*)(ak + (size_t)ar0 * F + aq0 * 8);
    ra1v = *(const uint4*)(ak + (size_t)ar1 * F + aq0 * 8);
#endif
  };
#if !defined(USE_TDM) && !defined(USE_ASYNC_LDS)
  auto stageA = [&](int p) {
    unsigned char* bufA = smem + p * BUF;
    *(uint4*)(bufA + ar0 * LDA + aq0 * 16) = ra0v;
    *(uint4*)(bufA + ar1 * LDA + aq0 * 16) = ra1v;
  };
#endif
  auto waitA = [&]() {
#if defined(USE_TDM)
    if (wave == 0) __builtin_amdgcn_s_wait_tensorcnt(0);
#elif defined(USE_ASYNC_LDS)
    __builtin_amdgcn_s_wait_asynccnt(0);
#endif
  };
  auto prefetch = [&](int kt) {
    const ushort_t* ak = Ge + (size_t)m0 * F + (size_t)kt * BK;
    __builtin_prefetch(ak + (size_t)ar0 * F + aq0 * 8, 0, 1);
    __builtin_prefetch(ak + (size_t)ar1 * F + aq0 * 8, 0, 1);
    const ushort_t* bk = W2e + (size_t)(kt * BK + kp * 2) * H + gc0;
    __builtin_prefetch(bk, 0, 1);
    __builtin_prefetch(bk + H, 0, 1);
  };
  auto stageB = [&](int p) {
    unsigned char* bufB = smem + p * BUF + ABYTES;
    const unsigned* r0 = (const unsigned*)&rb0;
    const unsigned* r1 = (const unsigned*)&rb1;
#pragma unroll
    for (int w = 0; w < 4; ++w) {
      unsigned lo = __builtin_amdgcn_perm(r1[w], r0[w], 0x05040100u);
      unsigned hi = __builtin_amdgcn_perm(r1[w], r0[w], 0x07060302u);
      *(unsigned*)(bufB + (c0 + 2 * w)     * LDB + kp * 4) = lo;
      *(unsigned*)(bufB + (c0 + 2 * w + 1) * LDB + kp * 4) = hi;
    }
  };

  constexpr int KT = F / BK;
  fetchB(0);
  loadA(0, 0);
#if !defined(USE_TDM) && !defined(USE_ASYNC_LDS)
  stageA(0);
#endif
  stageB(0);
  for (int kt = 0; kt < KT; ++kt) {
    const int p = kt & 1;
    const bool hasNext = (kt + 1) < KT;
    if (hasNext) fetchB(kt + 1);
#if !defined(USE_TDM) && !defined(USE_ASYNC_LDS)
    if (hasNext) loadA(kt + 1, p ^ 1);
#endif
    if (kt + 2 < KT) prefetch(kt + 2);
    waitA();
    __syncthreads();
    if (hasNext) {
      stageB(p ^ 1);
#if defined(USE_TDM) || defined(USE_ASYNC_LDS)
      loadA(kt + 1, p ^ 1);
#else
      stageA(p ^ 1);
#endif
    }

    const unsigned char* bufA = smem + p * BUF;
    const unsigned char* bufB = bufA + ABYTES;

    bf16x16 af[2], bf[4];
#pragma unroll
    for (int mi = 0; mi < 2; ++mi) {
      int row = wm * 32 + mi * 16 + l15;
      af[mi] = ldsFrag(bufA + row * LDA + hs * 16);
    }
#pragma unroll
    for (int ni = 0; ni < 4; ++ni) {
      int col = wn * 64 + ni * 16 + l15;
      bf[ni] = ldsFrag(bufB + col * LDB + hs * 16);
    }
#pragma unroll
    for (int mi = 0; mi < 2; ++mi)
#pragma unroll
      for (int ni = 0; ni < 4; ++ni)
        acc[mi][ni] = __builtin_amdgcn_wmma_f32_16x16x32_bf16(
            false, af[mi], false, bf[ni], (short)0, acc[mi][ni], false, false);
  }

#pragma unroll
  for (int mi = 0; mi < 2; ++mi)
#pragma unroll
    for (int ni = 0; ni < 4; ++ni) {
      int col = n0 + wn * 64 + ni * 16 + l15;
#pragma unroll
      for (int r = 0; r < 8; ++r) {
        int row = m0 + wm * 32 + mi * 16 + r + 8 * hs;
        Ye[(size_t)row * H + col] = f2bf(acc[mi][ni][r]);
      }
    }
}

extern "C" void kernel_launch(void* const* d_in, const int* in_sizes, int n_in,
                              void* d_out, int out_size, void* d_ws, size_t ws_size,
                              hipStream_t stream) {
  const ushort_t* x  = (const ushort_t*)d_in[0];   // [E, T, H] bf16
  const ushort_t* w1 = (const ushort_t*)d_in[1];   // [E*H, 2F] == [E][H][2F] bf16
  const ushort_t* w2 = (const ushort_t*)d_in[2];   // [E*F, H]  == [E][F][H]  bf16
  ushort_t*       y  = (ushort_t*)d_out;           // [E, T, H] bf16
  ushort_t*       g  = (ushort_t*)d_ws;            // [E, T, F] bf16 intermediate (64 MB)

  dim3 grid1(T / BM, F / 64, E);    // (16, 32, 8)
  dim3 grid2(T / BM, H / 128, E);   // (16, 32, 8)
  moe_fc1_glu<<<grid1, 256, 0, stream>>>(x, w1, g);
  moe_fc2<<<grid2, 256, 0, stream>>>(g, w2, y);
}
